// Attention_49289044688898
// MI455X (gfx1250) — matrix operand order repaired and checked by reading, not compiled, not run on hardware
//
#include <hip/hip_runtime.h>

typedef __bf16 bf16;
typedef __attribute__((ext_vector_type(16))) bf16  v16bf;
typedef __attribute__((ext_vector_type(8)))  bf16  v8bf;
typedef __attribute__((ext_vector_type(8)))  float v8f;

constexpr int BATCH = 512;   // B
constexpr int TT    = 64;    // encoder steps
constexpr int CC    = 512;   // encoder channels
constexpr int HH    = 512;   // hidden
constexpr int NCLS  = 96;    // classes
constexpr int STEPS = 26;    // batch_max_length + 1
constexpr int XW    = CC + NCLS + HH;  // 1120 = concat(context, onehot, h)
constexpr int ZW    = 4 * HH;          // 2048

__device__ __forceinline__ float sigmoidf_(float x) { return 1.0f / (1.0f + expf(-x)); }

// Async copy of 16 bytes global -> LDS via the gfx1250 ASYNC path (tracked by ASYNCcnt).
__device__ __forceinline__ void async_b128_to_lds(uint32_t lds_off, const void* gsrc) {
  asm volatile("global_load_async_to_lds_b128 %0, %1, off"
               :: "v"(lds_off), "v"(gsrc) : "memory");
}
__device__ __forceinline__ void wait_asynccnt0() {
  asm volatile("s_wait_asynccnt 0x0" ::: "memory");
}

// ---------- f32 -> bf16 elementwise ----------
__global__ void cvt_f32_bf16(const float* __restrict__ src, bf16* __restrict__ dst, size_t n) {
  size_t i = (size_t)blockIdx.x * blockDim.x + threadIdx.x;
  size_t stride = (size_t)gridDim.x * blockDim.x;
  for (; i < n; i += stride) dst[i] = (bf16)src[i];
}

// ---------- pack f32 weights [K x N] (optionally stacked W1 over W2) into WMMA B-fragment layout ----------
// P layout: [kt][nt][lane(32)][j(16)] with B-frag rule: frag[j] = W[kt*32 + (lane>=16 ? 16:0) + j][nt*16 + (lane&15)]
__global__ void pack_weights(const float* __restrict__ W1, int K1,
                             const float* __restrict__ W2, int K2,
                             int N, bf16* __restrict__ P) {
  const int K = K1 + K2;
  const size_t total = (size_t)K * N;
  size_t idx = (size_t)blockIdx.x * blockDim.x + threadIdx.x;
  if (idx >= total) return;
  const int ntiles = N >> 4;
  const size_t tile = idx >> 9;          // 512 elems per 32x16 tile
  const int r    = (int)(idx & 511);
  const int lane = r >> 4;
  const int j    = r & 15;
  const int kt = (int)(tile / ntiles);
  const int nt = (int)(tile % ntiles);
  const int n = nt * 16 + (lane & 15);
  const int k = kt * 32 + ((lane >> 4) * 16) + j;
  const float v = (k < K1) ? W1[(size_t)k * N + n] : W2[(size_t)(k - K1) * N + n];
  P[idx] = (bf16)v;
}

// ---------- WMMA bf16 GEMM: C[M,N] = A[M,K] * Bpacked[K,N] (+bias[N]) ----------
// Wave tile: 32 rows x 32 cols (2 A frags x 2 B frags -> 4 accumulators).
// Block = 8 waves stacked on M -> 256 x 32 per block. The 32-col B slice is shared by
// all 8 waves, so it is staged block-wide through LDS with double-buffered
// GLOBAL_LOAD_ASYNC_TO_LDS_B128 copies (chunks of 4 k-tiles = 8KB per buffer).
// NOTE: grids must divide M,N exactly (barriers inside => no early return).
template <typename OutT>
__global__ __launch_bounds__(256)
void wmma_gemm(const bf16* __restrict__ A, const bf16* __restrict__ Bp,
               OutT* __restrict__ C, const float* __restrict__ bias,
               int M, int N, int K, int lda, int ldc) {
  __shared__ alignas(16) bf16 smem[2 * 4096];     // 2 x 8KB double buffer
  const int tid   = threadIdx.x;
  const int lane  = tid & 31;
  const int wave  = tid >> 5;
  const int mBase = (blockIdx.y * 8 + wave) * 32;
  const int nBase = blockIdx.x * 32;
  const int ntiles = N >> 4;
  const int nt0    = nBase >> 4;
  const int nkt    = K >> 5;                       // 32-wide k-tiles
  // Low 32 bits of the flat shared address ARE the LDS offset (ISA 10.2 aperture rule).
  const uint32_t smbase = (uint32_t)(uintptr_t)(&smem[0]);

  // Issue async copies of one chunk (up to 4 k-tiles x 2 n-tiles) into buffer bufIdx.
  // Each k-tile slice is 2048B contiguous in the packed layout -> 128 lanes x 16B.
  auto issue_chunk = [&](int kt0, int bufIdx) {
    const int tiles = nkt - kt0 < 4 ? nkt - kt0 : 4;
    const int half  = tid >> 7;                    // threads 0-127 / 128-255
    const int slot  = tid & 127;
    for (int ktl = half; ktl < tiles; ktl += 2) {
      uint32_t ldst = smbase + (uint32_t)(bufIdx * 8192 + ktl * 2048 + slot * 16);
      const char* src = (const char*)(Bp + ((size_t)(kt0 + ktl) * ntiles + nt0) * 512) + slot * 16;
      async_b128_to_lds(ldst, src);
    }
  };

  // A fragment addressing: contiguous k order (k = 16*(lane/16) + s), same as packed B
  const int m0    = mBase + (lane & 15);
  const int kHalf = (lane >> 4) * 16;
  const bf16* arow0 = A + (size_t)m0 * lda + kHalf;
  const bf16* arow1 = arow0 + (size_t)16 * lda;
  v8f c00 = {}; v8f c01 = {}; v8f c10 = {}; v8f c11 = {};

  issue_chunk(0, 0);
  wait_asynccnt0();
  __syncthreads();

  const int nchunks = (nkt + 3) >> 2;
  for (int kc = 0; kc < nchunks; ++kc) {
    if (kc + 1 < nchunks) issue_chunk((kc + 1) * 4, (kc + 1) & 1);
    const int buf   = kc & 1;
    const int tiles = nkt - kc * 4 < 4 ? nkt - kc * 4 : 4;
    const bf16* bb  = smem + buf * 4096 + lane * 16;   // per-lane packed B fragments in LDS

    auto ktile = [&](int ktl) {
      const int k = (kc * 4 + ktl) * 32;
      v8bf a0lo = *(const v8bf*)(arow0 + k);        // K = kHalf + [0,8)
      v8bf a0hi = *(const v8bf*)(arow0 + k + 8);    // K = kHalf + 8 + [0,8)
      v8bf a1lo = *(const v8bf*)(arow1 + k);
      v8bf a1hi = *(const v8bf*)(arow1 + k + 8);
      v16bf a0 = __builtin_shufflevector(a0lo, a0hi, 0,1,2,3,4,5,6,7,8,9,10,11,12,13,14,15);
      v16bf a1 = __builtin_shufflevector(a1lo, a1hi, 0,1,2,3,4,5,6,7,8,9,10,11,12,13,14,15);
      v16bf b0 = *(const v16bf*)(bb + ktl * 1024);       // ds_load_b128 x2
      v16bf b1 = *(const v16bf*)(bb + ktl * 1024 + 512); // next 16-col tile
      c00 = __builtin_amdgcn_wmma_f32_16x16x32_bf16(false, a0, false, b0, (short)0, c00, false, false);
      c01 = __builtin_amdgcn_wmma_f32_16x16x32_bf16(false, a0, false, b1, (short)0, c01, false, false);
      c10 = __builtin_amdgcn_wmma_f32_16x16x32_bf16(false, a1, false, b0, (short)0, c10, false, false);
      c11 = __builtin_amdgcn_wmma_f32_16x16x32_bf16(false, a1, false, b1, (short)0, c11, false, false);
    };

    if (tiles == 4) {
      // fast path: fully unrolled so the scheduler pipelines loads across k-tiles
      #pragma unroll
      for (int ktl = 0; ktl < 4; ++ktl) ktile(ktl);
    } else {
      for (int ktl = 0; ktl < tiles; ++ktl) ktile(ktl);
    }

    if (kc + 1 < nchunks) {
      wait_asynccnt0();     // next chunk landed in LDS
      __syncthreads();      // all waves done reading current buffer + see new data
    }
  }

  // C/D layout: VGPR r -> row base + r (lanes 0-15) / base + 8 + r (lanes 16-31), col = lane&15
  const int row0 = mBase + (lane >> 4) * 8;
  const int col  = nBase + (lane & 15);
  const float bv0 = bias ? bias[col]      : 0.0f;
  const float bv1 = bias ? bias[col + 16] : 0.0f;
  #pragma unroll
  for (int r = 0; r < 8; ++r) {
    C[(size_t)(row0 + r) * ldc + col]           = (OutT)(c00[r] + bv0);
    C[(size_t)(row0 + r) * ldc + col + 16]      = (OutT)(c01[r] + bv1);
    C[(size_t)(row0 + 16 + r) * ldc + col]      = (OutT)(c10[r] + bv0);
    C[(size_t)(row0 + 16 + r) * ldc + col + 16] = (OutT)(c11[r] + bv1);
  }
}

// ---------- zero initial LSTM state ----------
__global__ void init_state(bf16* __restrict__ h_bf, float* __restrict__ c, bf16* __restrict__ xbuf) {
  const int b = blockIdx.x;
  for (int j = threadIdx.x; j < HH; j += blockDim.x) {
    h_bf[(size_t)b * HH + j] = (bf16)0.0f;
    c[(size_t)b * HH + j] = 0.0f;
    xbuf[(size_t)b * XW + CC + NCLS + j] = (bf16)0.0f;
  }
}

// ---------- attention: e = tanh(Hproj + q) @ Ws ; alpha = softmax_T(e) ; context = sum_t alpha * batch_H ----------
__global__ __launch_bounds__(256)
void attention_step(const bf16* __restrict__ Hproj, const float* __restrict__ q,
                    const float* __restrict__ batch_H, const float* __restrict__ Ws,
                    bf16* __restrict__ xbuf) {
  __shared__ float qs[HH];
  __shared__ float wss[HH];
  __shared__ float ered[TT];
  const int b = blockIdx.x;
  const int tid = threadIdx.x;
  const int wave = tid >> 5;
  const int lane = tid & 31;
  for (int i = tid; i < HH; i += 256) { qs[i] = q[(size_t)b * HH + i]; wss[i] = Ws[i]; }
  __syncthreads();
  // per-wave reduction: each wave owns 8 encoder steps
  for (int t = wave; t < TT; t += 8) {
    const bf16* hp = Hproj + ((size_t)b * TT + t) * HH;
    float acc = 0.0f;
    for (int i = lane; i < HH; i += 32) acc += tanhf((float)hp[i] + qs[i]) * wss[i];
    for (int off = 16; off > 0; off >>= 1) acc += __shfl_down(acc, off);
    if (lane == 0) ered[t] = acc;
  }
  __syncthreads();
  // softmax over T=64 using wave 0 (2 values per lane)
  if (tid < 32) {
    float e0 = ered[lane], e1 = ered[lane + 32];
    float mx = fmaxf(e0, e1);
    for (int off = 16; off > 0; off >>= 1) mx = fmaxf(mx, __shfl_xor(mx, off));
    float x0 = expf(e0 - mx), x1 = expf(e1 - mx);
    float sm = x0 + x1;
    for (int off = 16; off > 0; off >>= 1) sm += __shfl_xor(sm, off);
    ered[lane] = x0 / sm;
    ered[lane + 32] = x1 / sm;
  }
  __syncthreads();
  // context[c] = sum_t alpha[t] * batch_H[b,t,c] -> bf16 into x buffer
  for (int c0 = tid; c0 < CC; c0 += 256) {
    const float* bp = batch_H + (size_t)b * TT * CC + c0;
    float acc = 0.0f;
    #pragma unroll 8
    for (int t = 0; t < TT; ++t) acc += ered[t] * bp[(size_t)t * CC];
    xbuf[(size_t)b * XW + c0] = (bf16)acc;
  }
}

// ---------- write one-hot(char) into x buffer ----------
__global__ void set_onehot(const int* __restrict__ text, bf16* __restrict__ xbuf, int step) {
  const int b = blockIdx.x;
  const int j = threadIdx.x;            // NCLS threads
  const int tgt = text[(size_t)b * STEPS + step];
  xbuf[(size_t)b * XW + CC + j] = (bf16)(j == tgt ? 1.0f : 0.0f);
}

// ---------- LSTM gates: z already = x@Wcat + bias ----------
__global__ __launch_bounds__(256)
void lstm_cell(const float* __restrict__ z, float* __restrict__ c,
               bf16* __restrict__ h_bf, bf16* __restrict__ xbuf,
               bf16* __restrict__ hs, int step) {
  const int b = blockIdx.x;
  const float* zb = z + (size_t)b * ZW;
  for (int j = threadIdx.x; j < HH; j += blockDim.x) {
    const float iv = zb[j];
    const float fv = zb[HH + j];
    const float gv = zb[2 * HH + j];
    const float ov = zb[3 * HH + j];
    const float cp = c[(size_t)b * HH + j];
    const float cn = sigmoidf_(fv) * cp + sigmoidf_(iv) * tanhf(gv);
    const float hn = sigmoidf_(ov) * tanhf(cn);
    c[(size_t)b * HH + j] = cn;
    const bf16 hb = (bf16)hn;
    h_bf[(size_t)b * HH + j] = hb;                       // A matrix for next q-GEMM
    xbuf[(size_t)b * XW + CC + NCLS + j] = hb;           // h section of next x
    hs[((size_t)b * STEPS + step) * HH + j] = hb;        // saved for output GEMM
  }
}

extern "C" void kernel_launch(void* const* d_in, const int* in_sizes, int n_in,
                              void* d_out, int out_size, void* d_ws, size_t ws_size,
                              hipStream_t stream) {
  (void)in_sizes; (void)n_in; (void)out_size; (void)ws_size;
  const float* batch_H     = (const float*)d_in[0];
  const int*   text        = (const int*)d_in[1];
  // d_in[2] = batch_max_length (device scalar); problem size is static: STEPS = 26
  const float* Wi          = (const float*)d_in[3];
  const float* Wh          = (const float*)d_in[4];
  const float* bh          = (const float*)d_in[5];
  const float* Ws          = (const float*)d_in[6];
  const float* lstm_kernel = (const float*)d_in[7];
  const float* lstm_rec    = (const float*)d_in[8];
  const float* lstm_bias   = (const float*)d_in[9];
  const float* Wgen        = (const float*)d_in[10];
  const float* bgen        = (const float*)d_in[11];
  float* out = (float*)d_out;

  size_t off = 0;
  auto alloc = [&](size_t bytes) -> void* {
    off = (off + 255) & ~(size_t)255;
    void* p = (char*)d_ws + off;
    off += bytes;
    return p;
  };
  bf16*  bH_bf = (bf16*)alloc((size_t)BATCH * TT * CC * 2);
  bf16*  Hproj = (bf16*)alloc((size_t)BATCH * TT * HH * 2);
  bf16*  Wip   = (bf16*)alloc((size_t)CC * HH * 2);
  bf16*  Whp   = (bf16*)alloc((size_t)HH * HH * 2);
  bf16*  Wcatp = (bf16*)alloc((size_t)XW * ZW * 2);
  bf16*  Wgenp = (bf16*)alloc((size_t)HH * NCLS * 2);
  float* q     = (float*)alloc((size_t)BATCH * HH * 4);
  bf16*  xbuf  = (bf16*)alloc((size_t)BATCH * XW * 2);
  float* z     = (float*)alloc((size_t)BATCH * ZW * 4);
  bf16*  h_bf  = (bf16*)alloc((size_t)BATCH * HH * 2);
  float* cst   = (float*)alloc((size_t)BATCH * HH * 4);
  bf16*  hs    = (bf16*)alloc((size_t)BATCH * STEPS * HH * 2);

  // one-time conversions / packing
  cvt_f32_bf16<<<16384, 256, 0, stream>>>(batch_H, bH_bf, (size_t)BATCH * TT * CC);
  pack_weights<<<(CC * HH + 255) / 256, 256, 0, stream>>>(Wi, CC, nullptr, 0, HH, Wip);
  pack_weights<<<(HH * HH + 255) / 256, 256, 0, stream>>>(Wh, HH, nullptr, 0, HH, Whp);
  pack_weights<<<(XW * ZW + 255) / 256, 256, 0, stream>>>(lstm_kernel, CC + NCLS, lstm_rec, HH, ZW, Wcatp);
  pack_weights<<<(HH * NCLS + 255) / 256, 256, 0, stream>>>(Wgen, HH, nullptr, 0, NCLS, Wgenp);
  init_state<<<BATCH, 256, 0, stream>>>(h_bf, cst, xbuf);

  // Hproj = batch_H @ Wi  (M=32768, N=512, K=512), bf16 output (kept for 26x L2-resident re-reads)
  {
    dim3 g(HH / 32, (BATCH * TT) / 256);
    wmma_gemm<bf16><<<g, 256, 0, stream>>>(bH_bf, Wip, Hproj, nullptr, BATCH * TT, HH, CC, CC, HH);
  }

  for (int s = 0; s < STEPS; ++s) {
    // q = h @ Wh + bh
    dim3 gq(HH / 32, BATCH / 256);
    wmma_gemm<float><<<gq, 256, 0, stream>>>(h_bf, Whp, q, bh, BATCH, HH, HH, HH, HH);
    attention_step<<<BATCH, 256, 0, stream>>>(Hproj, q, batch_H, Ws, xbuf);
    set_onehot<<<BATCH, NCLS, 0, stream>>>(text, xbuf, s);
    // z = [context|onehot|h] @ [lstm_kernel; lstm_rec] + lstm_bias   (fused input+recurrent GEMM)
    dim3 gz(ZW / 32, BATCH / 256);
    wmma_gemm<float><<<gz, 256, 0, stream>>>(xbuf, Wcatp, z, lstm_bias, BATCH, ZW, XW, XW, ZW);
    lstm_cell<<<BATCH, 256, 0, stream>>>(z, cst, h_bf, xbuf, hs, s);
  }

  // probs = hs @ Wgen + bgen  (M=13312, N=96, K=512)
  {
    dim3 gp(NCLS / 32, (BATCH * STEPS) / 256);
    wmma_gemm<float><<<gp, 256, 0, stream>>>(hs, Wgenp, out, bgen, BATCH * STEPS, NCLS, HH, HH, NCLS);
  }
}
